// Transformer_block_16174846836970
// MI455X (gfx1250) — matrix operand order repaired and checked by reading, not compiled, not run on hardware
//
#include <hip/hip_runtime.h>

#define B_DIM    2
#define S_DIM    2048
#define D_MODEL  2048
#define NUM_HEADS 16
#define D_HEAD   128
#define D_FF     8192
#define M_ROWS   (B_DIM * S_DIM)   // 4096

typedef __attribute__((ext_vector_type(16))) __bf16 v16bf;
typedef __attribute__((ext_vector_type(8)))  __bf16 v8bf;
typedef __attribute__((ext_vector_type(4)))  __bf16 v4bf;
typedef __attribute__((ext_vector_type(8)))  float  v8f;
typedef __attribute__((ext_vector_type(4)))  unsigned int v4u;
typedef __attribute__((ext_vector_type(8)))  int v8i;
typedef __attribute__((ext_vector_type(4)))  int v4i;

typedef const __attribute__((address_space(3))) char* lds_cptr_t;

__device__ __forceinline__ unsigned lds_off_u32(const void* p) {
  // generic->LDS addrspacecast; ptr->size_t->u32 keeps the host pass legal
  return (unsigned)(size_t)(lds_cptr_t)p;
}

__device__ __forceinline__ v8f v8f_zero() {
  v8f z;
#pragma unroll
  for (int i = 0; i < 8; ++i) z[i] = 0.0f;
  return z;
}

__device__ __forceinline__ v8f wmma_bf16(v16bf a, v16bf b, v8f c) {
  return __builtin_amdgcn_wmma_f32_16x16x32_bf16(
      false, a, false, b, (short)0, c, false, false);
}

__device__ __forceinline__ v16bf pack16(v8bf lo, v8bf hi) {
  v16bf r;
#pragma unroll
  for (int i = 0; i < 8; ++i) { r[i] = lo[i]; r[i + 8] = hi[i]; }
  return r;
}

// --- CDNA5 async global->LDS copy (ASYNCcnt path) ------------------------
__device__ __forceinline__ void async_load_b128(unsigned lds_byte,
                                                const void* sbase, int voff) {
  asm volatile("global_load_async_to_lds_b128 %0, %1, %2"
               :
               : "v"(lds_byte), "v"(voff), "s"(sbase)
               : "memory");
}
__device__ __forceinline__ void wait_asynccnt0() {
  asm volatile("s_wait_asynccnt 0x0" ::: "memory");
}

// --- CDNA5 Tensor Data Mover: 2-D tile load (TENSORcnt path) -------------
__device__ __forceinline__ void tdm_load_2d(unsigned lds_byte, const void* gaddr,
                                            unsigned tensor_d0, unsigned tensor_d1,
                                            unsigned tile_d0, unsigned tile_d1,
                                            unsigned stride_d0) {
  const unsigned long long ga = (unsigned long long)(size_t)gaddr;
  v4u g0;
  g0[0] = 1u;                                       // count=1, user descriptor
  g0[1] = lds_byte;                                 // lds_addr
  g0[2] = (unsigned)(ga & 0xffffffffu);             // global_addr[31:0]
  g0[3] = (unsigned)((ga >> 32) & 0x1ffffffu) | (2u << 30);  // addr[56:32], type=2
  v8i g1;
  g1[0] = (int)(1u << 16);                          // data_size = 2 bytes
  g1[1] = (int)((tensor_d0 & 0xffffu) << 16);       // tensor_dim0[15:0]
  g1[2] = (int)((tensor_d0 >> 16) | ((tensor_d1 & 0xffffu) << 16));
  g1[3] = (int)((tensor_d1 >> 16) | (tile_d0 << 16));  // tile_dim0
  g1[4] = (int)(tile_d1 & 0xffffu);                 // tile_dim1 (tile_dim2=0)
  g1[5] = (int)stride_d0;                           // tensor_dim0_stride[31:0]
  g1[6] = 0;
  g1[7] = 0;
  const v4i z4 = {0, 0, 0, 0};
  const v8i z8 = {0, 0, 0, 0, 0, 0, 0, 0};
  __builtin_amdgcn_tensor_load_to_lds(g0, g1, z4, z4, z8, 0);
}

// ---------------------------------------------------------------------------
// Weight convert + transpose: in[K][N] fp32 -> out[N][K] bf16 (tiled, coalesced)
// ---------------------------------------------------------------------------
__global__ __launch_bounds__(256) void wtrans_kernel(const float* __restrict__ in,
                                                     __bf16* __restrict__ out,
                                                     int K, int N) {
  __shared__ float tile[32][33];
  const int k0 = blockIdx.y * 32;
  const int n0 = blockIdx.x * 32;
  const int tx = threadIdx.x & 31;
  const int ty = threadIdx.x >> 5;   // 0..7
  for (int i = ty; i < 32; i += 8)
    tile[i][tx] = in[(size_t)(k0 + i) * N + n0 + tx];
  __syncthreads();
  for (int i = ty; i < 32; i += 8)
    out[(size_t)(n0 + i) * K + k0 + tx] = (__bf16)tile[tx][i];
}

// ---------------------------------------------------------------------------
// RMSNorm: fp32 row in, bf16 row out
// ---------------------------------------------------------------------------
__global__ __launch_bounds__(256) void rmsnorm_kernel(const float* __restrict__ x,
                                                      const float* __restrict__ g,
                                                      __bf16* __restrict__ out,
                                                      int D) {
  __shared__ float red[256];
  const int row = blockIdx.x;
  const float* xr = x + (size_t)row * D;
  float ss = 0.0f;
  for (int i = threadIdx.x; i < D; i += 256) {
    float v = xr[i];
    ss += v * v;
  }
  red[threadIdx.x] = ss;
  __syncthreads();
  for (int s = 128; s > 0; s >>= 1) {
    if ((int)threadIdx.x < s) red[threadIdx.x] += red[threadIdx.x + s];
    __syncthreads();
  }
  const float rinv = rsqrtf(red[0] / (float)D + 1e-5f);
  for (int i = threadIdx.x; i < D; i += 256)
    out[(size_t)row * D + i] = (__bf16)(xr[i] * rinv * g[i]);
}

// ---------------------------------------------------------------------------
// RoPE + reshape [M, D] fp32 -> [B, H, S, dk] bf16
// ---------------------------------------------------------------------------
__global__ __launch_bounds__(256) void rope_reshape_kernel(const float* __restrict__ in,
                                                           __bf16* __restrict__ out,
                                                           int do_rope) {
  const size_t total = (size_t)M_ROWS * (D_MODEL / 2);
  size_t idx = (size_t)blockIdx.x * blockDim.x + threadIdx.x;
  if (idx >= total) return;
  const int row  = (int)(idx / (D_MODEL / 2));
  const int p    = (int)(idx % (D_MODEL / 2));
  const int hd   = p / (D_HEAD / 2);
  const int i    = p % (D_HEAD / 2);
  const int b    = row / S_DIM;
  const int spos = row % S_DIM;
  const float x1 = in[(size_t)row * D_MODEL + hd * D_HEAD + 2 * i];
  const float x2 = in[(size_t)row * D_MODEL + hd * D_HEAD + 2 * i + 1];
  float r1 = x1, r2 = x2;
  if (do_rope) {
    const float inv = __expf(-(2.0f * (float)i / (float)D_HEAD) * 9.210340371976184f);
    const float ang = (float)spos * inv;
    float sn, cs;
    __sincosf(ang, &sn, &cs);
    r1 = x1 * cs - x2 * sn;
    r2 = x1 * sn + x2 * cs;
  }
  const size_t ob = (((size_t)b * NUM_HEADS + hd) * S_DIM + spos) * D_HEAD + 2 * i;
  out[ob] = (__bf16)r1;
  out[ob + 1] = (__bf16)r2;
}

// ---------------------------------------------------------------------------
// bf16 WMMA GEMM: C[M,N] = A[M,K] * Wt[N,K]^T
// Double-buffered LDS; A tile via TDM, B tile via async global->LDS copies.
// EPI: 0 = fp32 store, 1 = bf16 store, 2 = fp32 store + fp32 residual
// ---------------------------------------------------------------------------
template <int EPI>
__global__ __launch_bounds__(256) void gemm_bf16_kernel(
    const __bf16* __restrict__ A, const __bf16* __restrict__ Wt,
    const float* __restrict__ R, void* __restrict__ Cout,
    int Mdim, int Ndim, int Kdim) {
  __shared__ alignas(32) __bf16 As[2][64][32];    // [buf][m][k]  4KB each
  __shared__ alignas(32) __bf16 Bs[2][128][32];   // [buf][n][k]  8KB each

  const int tid  = threadIdx.x;
  const int wave = tid >> 5;
  const int lane = tid & 31;
  const int wm = wave >> 2;            // 0..1
  const int wn = wave & 3;             // 0..3
  const int m0 = blockIdx.y * 64;
  const int n0 = blockIdx.x * 128;

  const int r16 = lane & 15;
  const int kb  = (lane < 16) ? 0 : 8;
  const int ks  = (lane < 16) ? 0 : 16;

  const unsigned asBase = lds_off_u32(&As[0][0][0]);
  const unsigned bsBase = lds_off_u32(&Bs[0][0][0]);

  v8f acc[2][2];
#pragma unroll
  for (int i = 0; i < 2; ++i)
#pragma unroll
    for (int j = 0; j < 2; ++j) acc[i][j] = v8f_zero();

  auto issue_tiles = [&](int k0, int buf) {
    // A tile 64x32 via TDM (one descriptor, wave 0 only; TDM ignores EXEC)
    if (tid < 32) {
      tdm_load_2d(asBase + (unsigned)buf * 4096,
                  A + (size_t)m0 * Kdim + k0,
                  (unsigned)Kdim, (unsigned)Mdim,
                  32u, 64u, (unsigned)Kdim);
    }
    // B tile 128x32 via per-lane async b128 copies (512 x 16B)
#pragma unroll
    for (int c = 0; c < 2; ++c) {
      const int idx = tid + c * 256;
      const int n  = idx >> 2;
      const int cb = (idx & 3) * 16;
      async_load_b128(bsBase + (unsigned)buf * 8192 + (unsigned)(n * 64 + cb),
                      Wt,
                      (int)(((size_t)(n0 + n) * Kdim + k0) * 2) + cb);
    }
  };

  const int nk = Kdim / 32;
  issue_tiles(0, 0);
  for (int it = 0; it < nk; ++it) {
    const int buf = it & 1;
    if (tid < 32) __builtin_amdgcn_s_wait_tensorcnt(0);
    wait_asynccnt0();
    __syncthreads();
    if (it + 1 < nk) issue_tiles((it + 1) * 32, buf ^ 1);

    v16bf af[2], bfr[2];
#pragma unroll
    for (int i = 0; i < 2; ++i) {
      const __bf16* p = &As[buf][wm * 32 + i * 16 + r16][kb];
      af[i]  = *(const v16bf*)(&As[buf][wm * 32 + i * 16 + r16][ks]);
      bfr[i] = *(const v16bf*)(&Bs[buf][wn * 32 + i * 16 + r16][ks]);
    }
#pragma unroll
    for (int i = 0; i < 2; ++i)
#pragma unroll
      for (int j = 0; j < 2; ++j)
        acc[i][j] = wmma_bf16(af[i], bfr[j], acc[i][j]);
  }

  const int rowoff = (lane < 16) ? 0 : 8;
#pragma unroll
  for (int i = 0; i < 2; ++i)
#pragma unroll
    for (int j = 0; j < 2; ++j)
#pragma unroll
      for (int v = 0; v < 8; ++v) {
        const int gr = m0 + wm * 32 + i * 16 + v + rowoff;
        const int gc = n0 + wn * 32 + j * 16 + r16;
        float val = acc[i][j][v];
        if (EPI == 2) val += R[(size_t)gr * Ndim + gc];
        if (EPI == 1)
          ((__bf16*)Cout)[(size_t)gr * Ndim + gc] = (__bf16)val;
        else
          ((float*)Cout)[(size_t)gr * Ndim + gc] = val;
      }
}

// ---------------------------------------------------------------------------
// Flash attention (causal, bf16 WMMA, fp32 online softmax)
// grid: (S/64, B*H), block 128 (4 waves). K tile staged async; V transposed.
// ---------------------------------------------------------------------------
__global__ __launch_bounds__(128) void flash_attn_kernel(
    const __bf16* __restrict__ Qh, const __bf16* __restrict__ Kh,
    const __bf16* __restrict__ Vh, __bf16* __restrict__ att) {
  __shared__ alignas(32) __bf16 Ks[32][D_HEAD];   // [kv][dk]
  __shared__ alignas(32) __bf16 Vs[D_HEAD][32];   // [dk][kv] (transposed)
  __shared__ alignas(32) __bf16 Ps[4][16][32];    // per-wave P staging

  const int bh = blockIdx.y;
  const int b  = bh / NUM_HEADS;
  const int hd = bh % NUM_HEADS;
  const int qb = blockIdx.x;
  const int tid  = threadIdx.x;
  const int wave = tid >> 5;
  const int lane = tid & 31;
  const int r16  = lane & 15;
  const int kb   = (lane < 16) ? 0 : 8;
  const int ks   = (lane < 16) ? 0 : 16;
  const int rowoff = (lane < 16) ? 0 : 8;

  const __bf16* Qb = Qh + (size_t)bh * S_DIM * D_HEAD;
  const __bf16* Kb = Kh + (size_t)bh * S_DIM * D_HEAD;
  const __bf16* Vb = Vh + (size_t)bh * S_DIM * D_HEAD;
  const unsigned ksBase = lds_off_u32(&Ks[0][0]);

  const int qrow = qb * 64 + wave * 16 + r16;

  v16bf qf[4];
  {
    const __bf16* qp = Qb + (size_t)qrow * D_HEAD;
#pragma unroll
    for (int kc = 0; kc < 4; ++kc)
      qf[kc] = *(const v16bf*)(qp + kc * 32 + ks);
  }

  v8f o[8];
  float mrow[8], lrow[8];
#pragma unroll
  for (int t = 0; t < 8; ++t) o[t] = v8f_zero();
#pragma unroll
  for (int v = 0; v < 8; ++v) { mrow[v] = -3.0e38f; lrow[v] = 0.0f; }

  const int nblk = 2 * qb + 2;
  for (int j = 0; j < nblk; ++j) {
    const int kvbase = j * 32;
    // K tile (32x128, 8KB) via async global->LDS: 512 x 16B chunks
    for (int c = tid; c < 512; c += 128) {
      const int kv = c >> 4;
      const int cb = (c & 15) * 16;
      async_load_b128(ksBase + (unsigned)(kv * 256 + cb),
                      Kb, (int)((kvbase + kv) * 256 + cb));
    }
    // V tile transposed into [dk][kv] via VGPR path
    for (int c = tid; c < 1024; c += 128) {
      const int kv  = c >> 5;
      const int dk4 = (c & 31) * 4;
      const v4bf vv = *(const v4bf*)(Vb + (size_t)(kvbase + kv) * D_HEAD + dk4);
#pragma unroll
      for (int i = 0; i < 4; ++i) Vs[dk4 + i][kv] = vv[i];
    }
    wait_asynccnt0();
    __syncthreads();

    v8f s0 = v8f_zero(), s1 = v8f_zero();
#pragma unroll
    for (int kc = 0; kc < 4; ++kc) {
      const v16bf b0 = *(const v16bf*)(&Ks[r16][kc * 32 + ks]);
      const v16bf b1 = *(const v16bf*)(&Ks[16 + r16][kc * 32 + ks]);
      s0 = wmma_bf16(qf[kc], b0, s0);
      s1 = wmma_bf16(qf[kc], b1, s1);
    }

    const float scale = 0.08838834764831845f;
    const int col0 = kvbase + r16;
    const int col1 = kvbase + 16 + r16;
    float corr[8];
#pragma unroll
    for (int v = 0; v < 8; ++v) {
      const int qpos = qb * 64 + wave * 16 + v + rowoff;
      float a0 = s0[v] * scale, a1 = s1[v] * scale;
      if (col0 > qpos) a0 = -3.0e38f;
      if (col1 > qpos) a1 = -3.0e38f;
      float tm = fmaxf(a0, a1);
#pragma unroll
      for (int off = 1; off < 16; off <<= 1)
        tm = fmaxf(tm, __shfl_xor(tm, off, 32));
      const float mnew = fmaxf(mrow[v], tm);
      corr[v] = __expf(mrow[v] - mnew);
      const float p0 = __expf(a0 - mnew);
      const float p1 = __expf(a1 - mnew);
      s0[v] = p0; s1[v] = p1;
      float ps = p0 + p1;
#pragma unroll
      for (int off = 1; off < 16; off <<= 1)
        ps += __shfl_xor(ps, off, 32);
      mrow[v] = mnew;
      lrow[v] = lrow[v] * corr[v] + ps;
    }
#pragma unroll
    for (int t = 0; t < 8; ++t)
#pragma unroll
      for (int v = 0; v < 8; ++v) o[t][v] *= corr[v];

#pragma unroll
    for (int v = 0; v < 8; ++v) {
      Ps[wave][v + rowoff][r16]      = (__bf16)s0[v];
      Ps[wave][v + rowoff][16 + r16] = (__bf16)s1[v];
    }
    const v16bf pf = *(const v16bf*)(&Ps[wave][r16][ks]);
#pragma unroll
    for (int t = 0; t < 8; ++t) {
      const v16bf vfr = *(const v16bf*)(&Vs[t * 16 + r16][ks]);
      o[t] = wmma_bf16(pf, vfr, o[t]);
    }
    __syncthreads();
  }

#pragma unroll
  for (int v = 0; v < 8; ++v) {
    const float inv = 1.0f / lrow[v];
    const int qpos = qb * 64 + wave * 16 + v + rowoff;
    const size_t rowbase =
        ((size_t)b * S_DIM + qpos) * D_MODEL + (size_t)hd * D_HEAD;
#pragma unroll
    for (int t = 0; t < 8; ++t)
      att[rowbase + t * 16 + r16] = (__bf16)(o[t][v] * inv);
  }
}

// ---------------------------------------------------------------------------
// SwiGLU: a <- silu(a) * b
// ---------------------------------------------------------------------------
__global__ __launch_bounds__(256) void swiglu_kernel(__bf16* __restrict__ a,
                                                     const __bf16* __restrict__ b,
                                                     size_t n) {
  size_t i = (size_t)blockIdx.x * blockDim.x + threadIdx.x;
  if (i < n) {
    const float av = (float)a[i];
    const float bv = (float)b[i];
    a[i] = (__bf16)(av / (1.0f + __expf(-av)) * bv);
  }
}

// ---------------------------------------------------------------------------
// host launch
// ---------------------------------------------------------------------------
extern "C" void kernel_launch(void* const* d_in, const int* in_sizes, int n_in,
                              void* d_out, int out_size, void* d_ws, size_t ws_size,
                              hipStream_t stream) {
  (void)in_sizes; (void)n_in; (void)out_size; (void)ws_size;
  const float* x  = (const float*)d_in[0];
  const float* wq = (const float*)d_in[1];
  const float* wk = (const float*)d_in[2];
  const float* wv = (const float*)d_in[3];
  const float* wo = (const float*)d_in[4];
  const float* g1 = (const float*)d_in[5];
  const float* g2 = (const float*)d_in[6];
  const float* w1 = (const float*)d_in[7];
  const float* w2 = (const float*)d_in[8];
  const float* w3 = (const float*)d_in[9];
  float* out = (float*)d_out;

  char* ws = (char*)d_ws;
  size_t off = 0;
  auto alloc = [&](size_t bytes) -> char* {
    char* p = ws + off;
    off += (bytes + 255) & ~(size_t)255;
    return p;
  };
  const size_t DD = (size_t)D_MODEL * D_MODEL;
  const size_t DF = (size_t)D_MODEL * D_FF;
  const size_t MD = (size_t)M_ROWS * D_MODEL;
  const size_t MF = (size_t)M_ROWS * D_FF;

  __bf16* wqb  = (__bf16*)alloc(DD * 2);   // transposed [N][K]
  __bf16* wkb  = (__bf16*)alloc(DD * 2);
  __bf16* wvb  = (__bf16*)alloc(DD * 2);
  __bf16* wob  = (__bf16*)alloc(DD * 2);
  __bf16* w1b  = (__bf16*)alloc(DF * 2);
  __bf16* w3b  = (__bf16*)alloc(DF * 2);
  __bf16* w2b  = (__bf16*)alloc(DF * 2);
  __bf16* hbuf = (__bf16*)alloc(MD * 2);
  float*  tmp  = (float*)alloc(MD * 4);
  __bf16* qhb  = (__bf16*)alloc(MD * 2);
  __bf16* khb  = (__bf16*)alloc(MD * 2);
  __bf16* vhb  = (__bf16*)alloc(MD * 2);
  __bf16* attb = (__bf16*)alloc(MD * 2);
  float*  out1 = (float*)alloc(MD * 4);
  __bf16* h2b  = (__bf16*)alloc(MD * 2);
  __bf16* abuf = (__bf16*)alloc(MF * 2);
  __bf16* bbuf = (__bf16*)alloc(MF * 2);

  // convert + transpose weights: [K][N] fp32 -> [N][K] bf16
  auto wt = [&](const float* src, __bf16* dst, int K, int N) {
    wtrans_kernel<<<dim3(N / 32, K / 32), 256, 0, stream>>>(src, dst, K, N);
  };
  wt(wq, wqb, D_MODEL, D_MODEL);
  wt(wk, wkb, D_MODEL, D_MODEL);
  wt(wv, wvb, D_MODEL, D_MODEL);
  wt(wo, wob, D_MODEL, D_MODEL);
  wt(w1, w1b, D_MODEL, D_FF);
  wt(w3, w3b, D_MODEL, D_FF);
  wt(w2, w2b, D_FF, D_MODEL);

  // h = rmsnorm(x, g1)
  rmsnorm_kernel<<<dim3(M_ROWS), dim3(256), 0, stream>>>(x, g1, hbuf, D_MODEL);

  const dim3 gD(D_MODEL / 128, M_ROWS / 64);
  const dim3 gF(D_FF / 128, M_ROWS / 64);
  const unsigned ropeBlocks = (unsigned)((MD / 2 + 255) / 256);

  gemm_bf16_kernel<0><<<gD, 256, 0, stream>>>(hbuf, wqb, nullptr, tmp, M_ROWS, D_MODEL, D_MODEL);
  rope_reshape_kernel<<<dim3(ropeBlocks), 256, 0, stream>>>(tmp, qhb, 1);
  gemm_bf16_kernel<0><<<gD, 256, 0, stream>>>(hbuf, wkb, nullptr, tmp, M_ROWS, D_MODEL, D_MODEL);
  rope_reshape_kernel<<<dim3(ropeBlocks), 256, 0, stream>>>(tmp, khb, 1);
  gemm_bf16_kernel<0><<<gD, 256, 0, stream>>>(hbuf, wvb, nullptr, tmp, M_ROWS, D_MODEL, D_MODEL);
  rope_reshape_kernel<<<dim3(ropeBlocks), 256, 0, stream>>>(tmp, vhb, 0);

  flash_attn_kernel<<<dim3(S_DIM / 64, B_DIM * NUM_HEADS), 128, 0, stream>>>(qhb, khb, vhb, attb);

  gemm_bf16_kernel<2><<<gD, 256, 0, stream>>>(attb, wob, x, out1, M_ROWS, D_MODEL, D_MODEL);
  rmsnorm_kernel<<<dim3(M_ROWS), dim3(256), 0, stream>>>(out1, g2, h2b, D_MODEL);

  gemm_bf16_kernel<1><<<gF, 256, 0, stream>>>(h2b, w1b, nullptr, abuf, M_ROWS, D_FF, D_MODEL);
  gemm_bf16_kernel<1><<<gF, 256, 0, stream>>>(h2b, w3b, nullptr, bbuf, M_ROWS, D_FF, D_MODEL);
  swiglu_kernel<<<dim3((unsigned)((MF + 255) / 256)), 256, 0, stream>>>(abuf, bbuf, MF);

  gemm_bf16_kernel<2><<<gD, 256, 0, stream>>>(abuf, w2b, out1, out, M_ROWS, D_MODEL, D_FF);
}
